// GNNModel_12094627905707
// MI455X (gfx1250) — matrix operand order repaired and checked by reading, not compiled, not run on hardware
//
#include <hip/hip_runtime.h>
#include <hip/hip_bf16.h>

// ---------------------------------------------------------------------------
// GNN forward for MI455X (gfx1250): bf16 WMMA GEMM engine + pointwise kernels.
// N=20000, IN=300, HID=512, DEG=8, HEADS=4, NCLS=23, 2 layers.
// All GEMM K-dims are zero-padded to multiples of 32 (both A and Bt), so the
// WMMA inner loop has no remainder guards at all.
// ---------------------------------------------------------------------------

#define NN     20000
#define IN_F   300
#define HID_F  512
#define DEG_N  8
#define HEADS_ 4
#define DH_    128
#define NCLS_  23

typedef __attribute__((ext_vector_type(16))) __bf16 v16bf;
typedef __attribute__((ext_vector_type(8)))  float  v8f;

static __device__ __forceinline__ float sigmf(float x) { return 1.f / (1.f + expf(-x)); }

// ---------------------------------------------------------------------------
// WMMA GEMM: C[nrows,M] = A[nrows,K] @ B[K,M] (+bias) (+Cin)
//   - K must be a multiple of 32; A cols [origK..K) and Bt cols [origK..K)
//     must be zero-padded by the caller.
//   - Bt is B transposed: [M,K] bf16, row stride ldb (>= K, multiple of 32)
//   - A is f32 (row stride lda, multiple of 32), converted to bf16 on the fly
//   - optional row gather: A-row for output row r is rowidx[r*idxStride+idxOff]
// Each wave computes a 16x64 tile: 4 x v_wmma_f32_16x16x32_bf16 per K-step.
// ---------------------------------------------------------------------------
__global__ void __launch_bounds__(256) gemm_bf16_16x64(
    const float* __restrict__ A, int lda,
    const __bf16* __restrict__ Bt, int ldb,
    const float* __restrict__ bias,
    const float* Cin, int ldcin,
    float* C, int ldc,
    int nrows, int K, int M,
    const int* __restrict__ rowidx, int idxStride, int idxOff)
{
  const int lane = threadIdx.x & 31;
  const int wave = threadIdx.x >> 5;
  const int mt = (M + 63) >> 6;
  const int rt = (nrows + 15) >> 4;
  const long tile = (long)blockIdx.x * 8 + wave;
  if (tile >= (long)rt * mt) return;            // wave-uniform exit
  const int rtile = (int)(tile / mt);
  const int ctile = (int)(tile - (long)rtile * mt);
  const int r0 = rtile << 4;
  const int c0 = ctile << 6;
  const int lm   = lane & 15;
  const int half = lane >> 4;

  int lrow = r0 + lm;
  if (lrow >= nrows) lrow = nrows - 1;          // clamp (stores are guarded)
  const int arow = rowidx ? rowidx[(long)lrow * idxStride + idxOff] : lrow;
  const float* __restrict__ Arow = A + (long)arow * lda + half * 8;

  // clamped B row pointers (clamped cols never stored; memory always valid)
  const __bf16* __restrict__ Brow[4];
#pragma unroll
  for (int j = 0; j < 4; ++j) {
    int col = c0 + j * 16 + lm;
    if (col >= M) col = M - 1;
    Brow[j] = Bt + (long)col * ldb + half * 8;
  }

  v8f zero = {};
  v8f acc[4];
#pragma unroll
  for (int j = 0; j < 4; ++j) acc[j] = zero;

  for (int k0 = 0; k0 < K; k0 += 32) {
    // A fragment (16-bit A 16x32 layout: two contiguous 8-runs per lane)
    v16bf a;
    const float* ap = Arow + k0;
#pragma unroll
    for (int e = 0; e < 8; ++e) a[e]     = (__bf16)ap[e];
#pragma unroll
    for (int e = 0; e < 8; ++e) a[8 + e] = (__bf16)ap[16 + e];
    // B fragments (same k order as A: two contiguous 8-runs per lane)
#pragma unroll
    for (int j = 0; j < 4; ++j) {
      const __bf16* __restrict__ bp = Brow[j] + k0;
      v16bf b;
#pragma unroll
      for (int e = 0; e < 16; ++e) b[e] = bp[(e & 7) + ((e >> 3) << 4)];
      acc[j] = __builtin_amdgcn_wmma_f32_16x16x32_bf16(
          false, a, false, b, (short)0, acc[j], false, false);
    }
  }

  // epilogue: C/D layout: elem v of lane L -> row r0+half*8+v, col c0+16j+lm
#pragma unroll
  for (int j = 0; j < 4; ++j) {
    const int col = c0 + j * 16 + lm;
    if (col >= M) continue;
    const float bv = bias ? bias[col] : 0.f;
#pragma unroll
    for (int v = 0; v < 8; ++v) {
      const int r = r0 + half * 8 + v;
      if (r >= nrows) continue;
      float val = acc[j][v] + bv;
      if (Cin) val += Cin[(long)r * ldcin + col];
      C[(long)r * ldc + col] = val;
    }
  }
}

// ---------------------------------------------------------------------------
// Row-wise LayerNorm over 512 features + optional activation.
// mode: 0=none, 1=relu, 2=elu.  One wave per row, shuffle reductions (wave32).
// ---------------------------------------------------------------------------
__global__ void __launch_bounds__(256) ln_rows(
    const float* X, int ldx,
    const float* __restrict__ g, const float* __restrict__ b,
    float* Y, int ldy, int nrows, int mode)
{
  const int lane = threadIdx.x & 31;
  const int wave = threadIdx.x >> 5;
  const int row = blockIdx.x * 8 + wave;
  if (row >= nrows) return;
  const float* x = X + (long)row * ldx;
  float v[16];
  float s = 0.f;
#pragma unroll
  for (int i = 0; i < 16; ++i) { v[i] = x[lane + i * 32]; s += v[i]; }
#pragma unroll
  for (int off = 16; off > 0; off >>= 1) s += __shfl_xor(s, off, 32);
  const float mean = s * (1.f / 512.f);
  float ss = 0.f;
#pragma unroll
  for (int i = 0; i < 16; ++i) { const float d = v[i] - mean; ss += d * d; }
#pragma unroll
  for (int off = 16; off > 0; off >>= 1) ss += __shfl_xor(ss, off, 32);
  const float inv = rsqrtf(ss * (1.f / 512.f) + 1e-12f);
  float* y = Y + (long)row * ldy;
#pragma unroll
  for (int i = 0; i < 16; ++i) {
    const int c = lane + i * 32;
    float o = g[c] * (v[i] - mean) * inv + b[c];
    if (mode == 1) o = fmaxf(o, 0.f);
    else if (mode == 2) o = (o > 0.f) ? o : expm1f(o);
    y[c] = o;
  }
}

// ---------------------------------------------------------------------------
// Small pointwise / prep kernels
// ---------------------------------------------------------------------------
__global__ void __launch_bounds__(256) k_zero(float* p, long n)
{
  const long i = (long)blockIdx.x * blockDim.x + threadIdx.x;
  if (i < n) p[i] = 0.f;
}

// Pack weight into Bt form [M, Kp] bf16, zero-padding k in [K, Kp).
// transpose=1: in is [K,M] (standard B), transpose=0: in is already [M,K].
__global__ void __launch_bounds__(256) k_pack_bt(const float* __restrict__ in,
                                                 __bf16* __restrict__ out,
                                                 int K, int M, int Kp, int transpose)
{
  const long i = (long)blockIdx.x * blockDim.x + threadIdx.x;
  const long total = (long)M * Kp;
  if (i >= total) return;
  const int m = (int)(i / Kp);
  const int k = (int)(i - (long)m * Kp);
  __bf16 v = (__bf16)0.f;
  if (k < K) v = (__bf16)(transpose ? in[(long)k * M + m] : in[(long)m * K + k]);
  out[i] = v;
}

__global__ void __launch_bounds__(256) k_vec_add(const float* __restrict__ a,
                                                 const float* __restrict__ b,
                                                 float* __restrict__ o, int n)
{
  const int i = blockIdx.x * blockDim.x + threadIdx.x;
  if (i < n) o[i] = a[i] + b[i];
}

// feat[n, 0:300]=embed[label[n]]; feat[n,300:600]=input_feats[n]; feat[n,600:608]=0
__global__ void __launch_bounds__(256) k_embed_concat(const float* __restrict__ feats,
                                                      const int* __restrict__ labels,
                                                      const float* __restrict__ embed,
                                                      float* __restrict__ feat)
{
  const long i = (long)blockIdx.x * blockDim.x + threadIdx.x;
  if (i >= (long)NN * 608) return;
  const int n = (int)(i / 608);
  const int j = (int)(i - (long)n * 608);
  float v = 0.f;
  if (j < IN_F)      v = embed[(long)labels[n] * IN_F + j];
  else if (j < 600)  v = feats[(long)n * IN_F + (j - IN_F)];
  feat[(long)n * 2048 + j] = v;
}

// PyTorch-order LSTM cell: gates = [i | f | g | o], each fin wide.
// h/c have row stride ldh (pad columns untouched, stay zero).
__global__ void __launch_bounds__(256) k_lstm_cell(const float* __restrict__ gates, int fin,
                                                   int ldh, float* h, float* c, long total)
{
  const long idx = (long)blockIdx.x * blockDim.x + threadIdx.x;
  if (idx >= total) return;
  const long n = idx / fin;
  const int  j = (int)(idx - n * fin);
  const float* g4 = gates + n * (4L * fin);
  const float ig = sigmf(g4[j]);
  const float fg = sigmf(g4[fin + j]);
  const float gg = tanhf(g4[2 * fin + j]);
  const float og = sigmf(g4[3 * fin + j]);
  const long o = n * ldh + j;
  const float cn = fg * c[o] + ig * gg;
  c[o] = cn;
  h[o] = og * tanhf(cn);
}

// es/ed per (node, head): dot(z[n, h*128:(h+1)*128], al/ar[h])
__global__ void __launch_bounds__(256) k_gat_scores(const float* __restrict__ z,
                                                    const float* __restrict__ al,
                                                    const float* __restrict__ ar,
                                                    float* __restrict__ es,
                                                    float* __restrict__ ed)
{
  const int idx = blockIdx.x * blockDim.x + threadIdx.x;
  if (idx >= NN * HEADS_) return;
  const int n = idx >> 2;
  const int h = idx & 3;
  const float* zr = z + (long)n * HID_F + h * DH_;
  const float* a = al + h * DH_;
  const float* r = ar + h * DH_;
  float sa = 0.f, sb = 0.f;
#pragma unroll 4
  for (int f = 0; f < DH_; ++f) { const float v = zr[f]; sa += v * a[f]; sb += v * r[f]; }
  es[idx] = sa;
  ed[idx] = sb;
}

// alpha[n,d,h] = softmax_d( leaky_relu(es[nbr[n,d],h] + ed[n,h], 0.2) )
__global__ void __launch_bounds__(256) k_gat_alpha(const float* __restrict__ es,
                                                   const float* __restrict__ ed,
                                                   const int* __restrict__ nbr,
                                                   float* __restrict__ alpha)
{
  const int idx = blockIdx.x * blockDim.x + threadIdx.x;
  if (idx >= NN * HEADS_) return;
  const int n = idx >> 2;
  const int h = idx & 3;
  const float e_d = ed[idx];
  float sc[DEG_N];
  float mx = -3.0e38f;
#pragma unroll
  for (int d = 0; d < DEG_N; ++d) {
    const int s = nbr[(long)n * DEG_N + d];
    float v = es[s * HEADS_ + h] + e_d;
    v = (v > 0.f) ? v : 0.2f * v;
    sc[d] = v;
    mx = fmaxf(mx, v);
  }
  float sum = 0.f;
#pragma unroll
  for (int d = 0; d < DEG_N; ++d) { sc[d] = expf(sc[d] - mx); sum += sc[d]; }
  const float r = 1.f / sum;
#pragma unroll
  for (int d = 0; d < DEG_N; ++d) alpha[((long)n * DEG_N + d) * HEADS_ + h] = sc[d] * r;
}

// h4raw[n,f] = sum_d alpha[n,d,f/128] * z[nbr[n,d], f] + gat_b[f]
__global__ void __launch_bounds__(256) k_gat_aggregate(const float* __restrict__ z,
                                                       const float* __restrict__ alpha,
                                                       const int* __restrict__ nbr,
                                                       const float* __restrict__ gb,
                                                       float* __restrict__ out)
{
  const long idx = (long)blockIdx.x * blockDim.x + threadIdx.x;
  if (idx >= (long)NN * HID_F) return;
  const int n = (int)(idx >> 9);
  const int f = (int)(idx & 511);
  const int h = f >> 7;
  float acc = 0.f;
#pragma unroll
  for (int d = 0; d < DEG_N; ++d) {
    const int s = nbr[(long)n * DEG_N + d];
    acc += alpha[((long)n * DEG_N + d) * HEADS_ + h] * z[(long)s * HID_F + f];
  }
  out[idx] = acc + gb[f];
}

// Deterministic BatchNorm stats, stage 1: per-block column partials (no atomics).
__global__ void __launch_bounds__(512) k_bn_partial(const float* __restrict__ y,
                                                    int nrows, int rowsPerBlock,
                                                    float* __restrict__ psum,
                                                    float* __restrict__ psumsq)
{
  const int c = threadIdx.x;           // 512 threads = 512 columns
  const int r0 = blockIdx.x * rowsPerBlock;
  int r1 = r0 + rowsPerBlock;
  if (r1 > nrows) r1 = nrows;
  float s = 0.f, ss = 0.f;
  for (int r = r0; r < r1; ++r) {
    const float v = y[(long)r * HID_F + c];
    s += v;
    ss += v * v;
  }
  psum[(long)blockIdx.x * HID_F + c] = s;
  psumsq[(long)blockIdx.x * HID_F + c] = ss;
}

__global__ void __launch_bounds__(512) k_bn_finalize(const float* __restrict__ psum,
                                                     const float* __restrict__ psumsq,
                                                     int nparts, float invN,
                                                     float* __restrict__ mean,
                                                     float* __restrict__ invstd)
{
  const int c = threadIdx.x;
  if (c >= HID_F) return;
  float s = 0.f, ss = 0.f;
  for (int p = 0; p < nparts; ++p) { s += psum[(long)p * HID_F + c]; ss += psumsq[(long)p * HID_F + c]; }
  const float m = s * invN;
  const float var = ss * invN - m * m;
  mean[c] = m;
  invstd[c] = rsqrtf(var + 1e-5f);
}

// gate = sigmoid(g*(y-mean)*invstd + be), in place over y
__global__ void __launch_bounds__(256) k_bn_gate(float* y,
                                                 const float* __restrict__ mean,
                                                 const float* __restrict__ invstd,
                                                 const float* __restrict__ g,
                                                 const float* __restrict__ be, long total)
{
  const long idx = (long)blockIdx.x * blockDim.x + threadIdx.x;
  if (idx >= total) return;
  const int c = (int)(idx & 511);
  const float v = g[c] * (y[idx] - mean[c]) * invstd[c] + be[c];
  y[idx] = sigmf(v);
}

// acc = (part==0 ? 0 : acc) + cat[:, part*512 + c] * w
__global__ void __launch_bounds__(256) k_se_accum(const float* __restrict__ cat,
                                                  const float* __restrict__ w,
                                                  float* acc, int part, long total)
{
  const long idx = (long)blockIdx.x * blockDim.x + threadIdx.x;
  if (idx >= total) return;
  const long r = idx >> 9;
  const int  c = (int)(idx & 511);
  const float v = cat[r * 1536 + part * HID_F + c] * w[idx];
  acc[idx] = (part == 0) ? v : (acc[idx] + v);
}

// ---------------------------------------------------------------------------
// Host orchestration
// ---------------------------------------------------------------------------
extern "C" void kernel_launch(void* const* d_in, const int* in_sizes, int n_in,
                              void* d_out, int out_size, void* d_ws, size_t ws_size,
                              hipStream_t stream)
{
  (void)in_sizes; (void)n_in; (void)out_size; (void)ws_size;

  const float* in_feats = (const float*)d_in[0];
  const int*   labels   = (const int*)d_in[1];
  const int*   nbr      = (const int*)d_in[2];
  auto P = [&](int i) { return (const float*)d_in[i]; };
  // params flattened in dict insertion order:
  const float* embed = P(3);
  const float* fmW1 = P(4);  const float* fmb1 = P(5);
  const float* fmlng = P(6); const float* fmlnb = P(7);
  const float* fmW2 = P(8);  const float* fmb2 = P(9);
  const int LBASE = 10, LSTRIDE = 33;
  const int HBASE = LBASE + 2 * LSTRIDE;                       // 76
  const float* hdW1 = P(HBASE + 0); const float* hdb1 = P(HBASE + 1);
  const float* hdlng = P(HBASE + 2); const float* hdlnb = P(HBASE + 3);
  const float* hdW2 = P(HBASE + 4); const float* hdb2 = P(HBASE + 5);

  // ---- workspace bump allocator ----
  char* base = (char*)d_ws;
  size_t off = 0;
  auto take = [&](size_t bytes) -> void* {
    void* p = base + off;
    off += (bytes + 255) & ~(size_t)255;
    return p;
  };

  __bf16* warena = (__bf16*)take(sizeof(__bf16) * 10600000ull);
  float* big    = (float*)take(sizeof(float) * (size_t)NN * 2048);  // feat / gates / z,y scratch
  float* hbuf   = (float*)take(sizeof(float) * (size_t)NN * 512);
  float* cat    = (float*)take(sizeof(float) * (size_t)NN * 1536);
  float* hstate = (float*)take(sizeof(float) * (size_t)NN * 512);
  float* cstate = (float*)take(sizeof(float) * (size_t)NN * 512);   // also SE fusion accumulator
  float* zbuf   = (float*)take(sizeof(float) * (size_t)NN * 512);
  float* es     = (float*)take(sizeof(float) * (size_t)NN * HEADS_);
  float* ed     = (float*)take(sizeof(float) * (size_t)NN * HEADS_);
  float* alpha  = (float*)take(sizeof(float) * (size_t)NN * DEG_N * HEADS_);
  float* psum   = (float*)take(sizeof(float) * 200 * 512);
  float* psumsq = (float*)take(sizeof(float) * 200 * 512);
  float* bmean  = (float*)take(sizeof(float) * 512);
  float* binv   = (float*)take(sizeof(float) * 512);
  float* bias0  = (float*)take(sizeof(float) * 1200);
  float* bias1  = (float*)take(sizeof(float) * 2048);

  const long NB = 256;
  auto gblk = [&](long n) { return dim3((unsigned)((n + NB - 1) / NB)); };
  auto kp = [](int K) { return (K + 31) & ~31; };

  // ---- weight prep: pack to Bt [M, Kp] bf16 with zero K-padding ----
  size_t woff = 0;
  auto pack = [&](const float* W, int K, int M, int transpose) -> const __bf16* {
    __bf16* dst = warena + woff;
    const int Kp = kp(K);
    woff += (size_t)M * Kp;
    k_pack_bt<<<gblk((long)M * Kp), NB, 0, stream>>>(W, dst, K, M, Kp, transpose);
    return dst;
  };

  const __bf16* fmW1t = pack(fmW1, 600, HID_F, 1);   // [512, 608]
  const __bf16* fmW2t = pack(fmW2, HID_F, IN_F, 1);  // [300, 512]
  struct LayerW {
    const __bf16 *skip, *wih, *whh, *selfw, *neigh, *gatw, *se[3];
    const float* biascomb;
  } LW[2];
  for (int l = 0; l < 2; ++l) {
    const int fin = (l == 0) ? IN_F : HID_F;
    const int b = LBASE + LSTRIDE * l;
    LW[l].skip  = pack(P(b + 0), fin, HID_F, 1);
    LW[l].wih   = pack(P(b + 2), fin, 4 * fin, 0);   // [4fin, fin] already Bt
    LW[l].whh   = pack(P(b + 3), fin, 4 * fin, 0);
    LW[l].selfw = pack(P(b + 6), fin, HID_F, 1);
    LW[l].neigh = pack(P(b + 7), fin, HID_F, 1);
    LW[l].gatw  = pack(P(b + 9), fin, HID_F, 1);
    LW[l].se[0] = pack(P(b + 21), 3 * HID_F, HID_F, 1);
    LW[l].se[1] = pack(P(b + 25), 3 * HID_F, HID_F, 1);
    LW[l].se[2] = pack(P(b + 29), 3 * HID_F, HID_F, 1);
    float* bc = (l == 0) ? bias0 : bias1;
    k_vec_add<<<gblk(4 * fin), NB, 0, stream>>>(P(b + 4), P(b + 5), bc, 4 * fin);
    LW[l].biascomb = bc;
  }
  const __bf16* hdW1t = pack(hdW1, HID_F, HID_F, 1);
  const __bf16* hdW2t = pack(hdW2, HID_F, NCLS_, 1);

  // ---- launch helpers ----
  auto gemm = [&](const float* A, int lda, const __bf16* Bt, int ldb,
                  const float* bias, const float* Cin, int ldcin,
                  float* C, int ldc, int nrows, int K, int M,
                  const int* rowidx, int idxStride, int idxOff) {
    const long rt = (nrows + 15) >> 4;
    const long mt = (M + 63) >> 6;
    const long tiles = rt * mt;
    gemm_bf16_16x64<<<dim3((unsigned)((tiles + 7) / 8)), 256, 0, stream>>>(
        A, lda, Bt, ldb, bias, Cin, ldcin, C, ldc, nrows, K, M, rowidx, idxStride, idxOff);
  };
  auto ln = [&](const float* X, int ldx, const float* g, const float* bb,
                float* Y, int ldy, int mode) {
    ln_rows<<<dim3((NN + 7) / 8), 256, 0, stream>>>(X, ldx, g, bb, Y, ldy, NN, mode);
  };
  auto zero = [&](float* p, long n) { k_zero<<<gblk(n), NB, 0, stream>>>(p, n); };

  // =========================== feature MLP ================================
  // feat = [embed[labels] | input_feats | 0pad]  (lda = 2048, inside `big`)
  k_embed_concat<<<gblk((long)NN * 608), NB, 0, stream>>>(in_feats, labels, embed, big);
  // y1 = feat @ W1 + b1   (stored at big cols 640..1151; 640*4B keeps 128B align)
  gemm(big, 2048, fmW1t, 608, fmb1, nullptr, 0, big + 640, 2048, NN, 608, HID_F,
       nullptr, 0, 0);
  ln(big + 640, 2048, fmlng, fmlnb, big + 640, 2048, /*relu*/1);
  // h = x @ W2 + b2  -> hbuf [N, 300] with ld 320, pad cols zeroed
  zero(hbuf, (long)NN * 320);
  gemm(big + 640, 2048, fmW2t, HID_F, fmb2, nullptr, 0, hbuf, 320, NN, HID_F, IN_F,
       nullptr, 0, 0);

  int hld = 320;                                  // current h: [N, fin], ld = kp(fin)
  for (int l = 0; l < 2; ++l) {
    const int fin = (l == 0) ? IN_F : HID_F;
    const int finp = kp(fin);
    const int b = LBASE + LSTRIDE * l;

    // ---- branch 1: h1 = elu(LN(h @ skip_W + skip_b)) -> cat[:, 0:512]
    gemm(hbuf, hld, LW[l].skip, finp, P(b + 1), nullptr, 0, zbuf, HID_F, NN, finp, HID_F,
         nullptr, 0, 0);
    ln(zbuf, HID_F, P(b + 13), P(b + 14), cat + 0, 1536, /*elu*/2);

    // ---- branch 2: LSTM over 8 gathered neighbors, then SAGE ----
    zero(hstate, (long)NN * finp);
    zero(cstate, (long)NN * finp);
    for (int t = 0; t < DEG_N; ++t) {
      // gates = h[nbr[:,t]] @ Wih^T + (bih+bhh)   (gather folded into A loads)
      gemm(hbuf, hld, LW[l].wih, finp, LW[l].biascomb, nullptr, 0, big, 4 * fin,
           NN, finp, 4 * fin, nbr, DEG_N, t);
      if (t > 0)  // gates += hstate @ Whh^T
        gemm(hstate, finp, LW[l].whh, finp, nullptr, big, 4 * fin, big, 4 * fin,
             NN, finp, 4 * fin, nullptr, 0, 0);
      k_lstm_cell<<<gblk((long)NN * fin), NB, 0, stream>>>(big, fin, finp, hstate, cstate,
                                                           (long)NN * fin);
    }
    // h2 = elu(LN(h @ self_W + hN @ neigh_W + sage_b)) -> cat[:, 512:1024]
    gemm(hbuf, hld, LW[l].selfw, finp, P(b + 8), nullptr, 0, zbuf, HID_F, NN, finp, HID_F,
         nullptr, 0, 0);
    gemm(hstate, finp, LW[l].neigh, finp, nullptr, zbuf, HID_F, zbuf, HID_F, NN, finp, HID_F,
         nullptr, 0, 0);
    ln(zbuf, HID_F, P(b + 15), P(b + 16), cat + 512, 1536, /*elu*/2);

    // ---- branch 3: multi-head GAT ----
    gemm(hbuf, hld, LW[l].gatw, finp, nullptr, nullptr, 0, zbuf, HID_F, NN, finp, HID_F,
         nullptr, 0, 0);                                    // z
    k_gat_scores<<<gblk((long)NN * HEADS_), NB, 0, stream>>>(zbuf, P(b + 10), P(b + 11), es, ed);
    k_gat_alpha<<<gblk((long)NN * HEADS_), NB, 0, stream>>>(es, ed, nbr, alpha);
    k_gat_aggregate<<<gblk((long)NN * HID_F), NB, 0, stream>>>(zbuf, alpha, nbr, P(b + 12), big);
    ln(big, HID_F, P(b + 17), P(b + 18), cat + 1024, 1536, /*elu*/2);

    // ---- SE-gated fusion (deterministic train-mode BatchNorm) ----
    float* ybuf = big + (size_t)NN * 512;  // disjoint flat region of `big`
    for (int i = 0; i < 3; ++i) {
      const int pb = b + 21 + 4 * i;
      gemm(cat, 1536, LW[l].se[i], 1536, P(pb + 1), nullptr, 0, ybuf, HID_F,
           NN, 1536, HID_F, nullptr, 0, 0);
      k_bn_partial<<<dim3(200), 512, 0, stream>>>(ybuf, NN, 100, psum, psumsq);
      k_bn_finalize<<<dim3(1), 512, 0, stream>>>(psum, psumsq, 200, 1.f / NN, bmean, binv);
      k_bn_gate<<<gblk((long)NN * HID_F), NB, 0, stream>>>(ybuf, bmean, binv,
                                                           P(pb + 2), P(pb + 3),
                                                           (long)NN * HID_F);
      k_se_accum<<<gblk((long)NN * HID_F), NB, 0, stream>>>(cat, ybuf, cstate, i,
                                                            (long)NN * HID_F);
    }
    // h = LN(h1*w1 + h2*w2 + h4*w3)  -> hbuf [N,512]
    ln(cstate, HID_F, P(b + 19), P(b + 20), hbuf, HID_F, /*none*/0);
    hld = HID_F;
  }

  // =============================== head ===================================
  gemm(hbuf, HID_F, hdW1t, HID_F, hdb1, nullptr, 0, zbuf, HID_F, NN, HID_F, HID_F,
       nullptr, 0, 0);
  ln(zbuf, HID_F, hdlng, hdlnb, zbuf, HID_F, /*relu*/1);
  gemm(zbuf, HID_F, hdW2t, HID_F, hdb2, nullptr, 0, (float*)d_out, NCLS_, NN, HID_F, NCLS_,
       nullptr, 0, 0);
}
